// SVMModel_24352464569965
// MI455X (gfx1250) — matrix operand order repaired and checked by reading, not compiled, not run on hardware
//
#include <hip/hip_runtime.h>
#include <hip/hip_bf16.h>

typedef __attribute__((ext_vector_type(16))) _Float16 v16h;
typedef __attribute__((ext_vector_type(8)))  float    v8f;

#define N_ROWS   16384
#define D_TS     1022
#define D_FULL   1024
#define N_CLASS  2
#define GAMMA    0.1f

#define ROW_BLOCKS   32
#define ROWS_PER_RB  (N_ROWS / ROW_BLOCKS)   // 512

// x[i, d] = d < 1022 ? ts[i*1022 + d] : (d == 1022 ? meta0[i] : meta1[i])
__device__ __forceinline__ float load_x(const float* __restrict__ ts,
                                        const float* __restrict__ m0,
                                        const float* __restrict__ m1,
                                        int r, int d) {
    if (d < D_TS) return ts[(size_t)r * D_TS + d];
    return (d == D_TS) ? m0[r] : m1[r];
}

__device__ __forceinline__ float feat_of(float x, float cs, float cq) {
    // sum_j (x[i,d]-x[j,d])^2 = N*x^2 - 2*x*colsum + colsqsum ; feat = exp(-g*.)
    const float ps = fmaf((float)N_ROWS * x, x, fmaf(-2.0f * x, cs, cq));
    return __expf(-GAMMA * ps);
}

// ---------------------------------------------------------------------------
// Pass 1a: per-row-block partial column sums / sq-sums.
// Thread t owns column d; loads are coalesced across the block (consecutive
// lanes -> consecutive addresses). This pass streams the 64 MB from HBM once.
// ---------------------------------------------------------------------------
__global__ void colsum_partial_kernel(const float* __restrict__ ts,
                                      const float* __restrict__ m0,
                                      const float* __restrict__ m1,
                                      float* __restrict__ partial) {
    const int d  = blockIdx.x * 256 + threadIdx.x;   // 0..1023
    const int rb = blockIdx.y;                        // 0..31
    const int r0 = rb * ROWS_PER_RB;

    float s = 0.0f, s2 = 0.0f;
    if (d < D_TS) {
        const float* __restrict__ col = ts + d;
        for (int r = r0; r < r0 + ROWS_PER_RB; ++r) {
            if (r + 16 < r0 + ROWS_PER_RB) {
                __builtin_prefetch(&col[(size_t)(r + 16) * D_TS], 0, 0); // global_prefetch_b8
            }
            const float v = col[(size_t)r * D_TS];
            s  += v;
            s2  = fmaf(v, v, s2);
        }
    } else {
        const float* __restrict__ col = (d == D_TS) ? m0 : m1;
        for (int r = r0; r < r0 + ROWS_PER_RB; ++r) {
            const float v = col[r];
            s  += v;
            s2  = fmaf(v, v, s2);
        }
    }
    partial[(size_t)rb * 2048 + d]        = s;
    partial[(size_t)rb * 2048 + 1024 + d] = s2;
}

// ---------------------------------------------------------------------------
// Pass 1b: deterministic fixed-order reduction of the 32 partials per column.
// ---------------------------------------------------------------------------
__global__ void colsum_reduce_kernel(const float* __restrict__ partial,
                                     float* __restrict__ colsum,
                                     float* __restrict__ colsq) {
    const int d = blockIdx.x * 256 + threadIdx.x;    // 0..1023
    float s = 0.0f, s2 = 0.0f;
    for (int rb = 0; rb < ROW_BLOCKS; ++rb) {
        s  += partial[(size_t)rb * 2048 + d];
        s2 += partial[(size_t)rb * 2048 + 1024 + d];
    }
    colsum[d] = s;
    colsq[d]  = s2;
}

// ---------------------------------------------------------------------------
// Pass 2: fused feat + GEMV via v_wmma_f32_16x16x32_f16.
// One wave32 owns a 16-row tile. Operand roles:
//   A (16x32 f16)  = svm_w  : M = class. Rows M>=2 are garbage (lane loads
//                             w[(m&1)*1024+..] unconditionally) -- harmless,
//                             since only D rows 0..1 are ever stored.
//   B (32x16 f16)  = feat^T : lane n owns output row n; its 16 halfs are 16
//                             consecutive K of that row -> 4x global_load_b128.
//   D (16x16 f32)  : acc[0] lanes 0-15 = class-0 logits, acc[1] = class-1.
// Main loop (ks=0..960) is branch-free; the tail k-step handles meta columns.
// x is L2-resident after pass 1 (64 MB < 192 MB L2).
// ---------------------------------------------------------------------------
__global__ void svm_feat_wmma_kernel(const float* __restrict__ ts,
                                     const float* __restrict__ m0,
                                     const float* __restrict__ m1,
                                     const float* __restrict__ w,      // [2,1024]
                                     const float* __restrict__ bias,   // [2]
                                     const float* __restrict__ colsum, // [1024]
                                     const float* __restrict__ colsq,  // [1024]
                                     float* __restrict__ out) {        // [N,2]
    const int lane  = threadIdx.x & 31;
    const int wave  = threadIdx.x >> 5;
    const int tile  = blockIdx.x * (blockDim.x >> 5) + wave;  // 0..1023
    const int row0  = tile * 16;
    const int h     = lane >> 4;      // K-half select (0 or 1)
    const int m     = lane & 15;      // A: M index / B: N (row-in-tile) index
    const int myrow = row0 + m;

    const float* __restrict__ rowp = ts + (size_t)myrow * D_TS;
    const float* __restrict__ wrow = w + (size_t)(m & 1) * D_FULL;

    v8f acc = {};

    // ---- Main loop: 31 k-steps, all columns < 992 < D_TS, no conditionals.
    for (int ks = 0; ks < 992; ks += 32) {
        const int kb = ks + 16 * h;                 // this lane's 16 B columns
        v16h bmat;
#pragma unroll
        for (int j = 0; j < 16; ++j) {
            const int d = kb + j;
            bmat[j] = (_Float16)feat_of(rowp[d], colsum[d], colsq[d]);
        }
        v16h amat;                                   // A-layout K(t)=16h+t (matches B)
#pragma unroll
        for (int t = 0; t < 16; ++t) {
            const int K = 16 * h + t;
            amat[t] = (_Float16)wrow[ks + K];
        }
        acc = __builtin_amdgcn_wmma_f32_16x16x32_f16(
            false, amat, false, bmat, (short)0, acc, false, false);
    }

    // ---- Tail k-step (ks = 992): columns 992..1023 include the meta columns.
    {
        const int ks = 992;
        const int kb = ks + 16 * h;
        v16h bmat;
#pragma unroll
        for (int j = 0; j < 16; ++j) {
            const int d   = kb + j;
            const float x = load_x(ts, m0, m1, myrow, d);
            bmat[j] = (_Float16)feat_of(x, colsum[d], colsq[d]);
        }
        v16h amat;
#pragma unroll
        for (int t = 0; t < 16; ++t) {
            const int K = 16 * h + t;
            amat[t] = (_Float16)wrow[ks + K];
        }
        acc = __builtin_amdgcn_wmma_f32_16x16x32_f16(
            false, amat, false, bmat, (short)0, acc, false, false);
    }

    // D: VGPR v, lanes 0-15 hold (M=v, N=lane) -> lane n stores row row0+n.
    if (lane < 16) {
        const size_t o = (size_t)(row0 + lane) * N_CLASS;
        out[o + 0] = acc[0] + bias[0];
        out[o + 1] = acc[1] + bias[1];
    }
}

// ---------------------------------------------------------------------------
extern "C" void kernel_launch(void* const* d_in, const int* in_sizes, int n_in,
                              void* d_out, int out_size, void* d_ws, size_t ws_size,
                              hipStream_t stream) {
    const float* ts   = (const float*)d_in[0];   // [16384,1022]
    const float* m0   = (const float*)d_in[1];   // [16384]
    const float* m1   = (const float*)d_in[2];   // [16384]
    const float* w    = (const float*)d_in[3];   // [2,1024]
    const float* bias = (const float*)d_in[4];   // [2]
    float*       out  = (float*)d_out;           // [16384,2]

    float* ws      = (float*)d_ws;
    float* colsum  = ws;                 // 1024 floats
    float* colsq   = ws + 1024;          // 1024 floats
    float* partial = ws + 2048;          // 32 * 2048 floats = 256 KB

    // Pass 1a: streaming partial column sums (coalesced, HBM-bound)
    colsum_partial_kernel<<<dim3(D_FULL / 256, ROW_BLOCKS), 256, 0, stream>>>(
        ts, m0, m1, partial);

    // Pass 1b: deterministic fixed-order serial reduce
    colsum_reduce_kernel<<<dim3(D_FULL / 256), 256, 0, stream>>>(
        partial, colsum, colsq);

    // Pass 2: fused feat + WMMA GEMV. 1024 row-tiles, 8 waves/block.
    const int tiles = N_ROWS / 16;                 // 1024
    const int waves_per_block = 8;                 // 256 threads
    svm_feat_wmma_kernel<<<dim3(tiles / waves_per_block), 32 * waves_per_block, 0, stream>>>(
        ts, m0, m1, w, bias, colsum, colsq, out);
}
